// Net_62629213110511
// MI455X (gfx1250) — matrix operand order repaired and checked by reading, not compiled, not run on hardware
//
#include <hip/hip_runtime.h>
#include <hip/hip_bf16.h>
#include <math.h>

typedef __attribute__((ext_vector_type(16))) _Float16 v16h;
typedef __attribute__((ext_vector_type(8)))  float    v8f;

#define NALU_EPS 1e-10f
#define BN_EPS   1e-5f

struct alignas(16) U128 { unsigned int x, y, z, w; };
union HF { U128 q[2]; v16h h; };

__device__ __forceinline__ float sigm_(float x) { return 1.0f / (1.0f + __expf(-x)); }

// ---------------------------------------------------------------- utilities
__global__ void k_zero(unsigned int* p, long n) {
  long i = (long)blockIdx.x * blockDim.x + threadIdx.x;
  long stride = (long)gridDim.x * blockDim.x;
  for (; i < n; i += stride) p[i] = 0u;
}

// W = tanh(What)*sigmoid(Mhat) and G, packed f16 [Opad, Kpad] (zero padded)
__global__ void k_pack_nalu_w(const float* __restrict__ What, const float* __restrict__ Mhat,
                              const float* __restrict__ G,
                              _Float16* __restrict__ Wd, _Float16* __restrict__ Gd,
                              int O, int K, int Opad, int Kpad) {
  int idx = blockIdx.x * blockDim.x + threadIdx.x;
  if (idx >= Opad * Kpad) return;
  int o = idx / Kpad, k = idx - o * Kpad;
  float w = 0.0f, g = 0.0f;
  if (o < O && k < K) {
    int s = o * K + k;
    w = tanhf(What[s]) * sigm_(Mhat[s]);
    g = G[s];
  }
  Wd[idx] = (_Float16)w;
  Gd[idx] = (_Float16)g;
}

// f32 [O,K] row-major -> f16 [Opad,Kpad], optional log(|x|+eps)
__global__ void k_pack_f16(const float* __restrict__ src, _Float16* __restrict__ dst,
                           int O, int K, int Opad, int Kpad, int logmode) {
  int idx = blockIdx.x * blockDim.x + threadIdx.x;
  if (idx >= Opad * Kpad) return;
  int o = idx / Kpad, k = idx - o * Kpad;
  float v = 0.0f;
  if (o < O && k < K) {
    v = src[o * K + k];
    if (logmode) v = __logf(fabsf(v) + NALU_EPS);
  }
  dst[idx] = (_Float16)v;
}

// ---------------------------------------------------------------- WMMA GEMM
// C[M,Npad] = A[M,Kpad] * Bt[Npad,Kpad]^T, f16 in / f32 out.
// One wave -> 16 x (16*NSUB) tile. Requires Kpad%32==0, M%16==0, Npad%(16*NSUB)==0.
template <int NSUB>
__global__ void k_gemm(const _Float16* __restrict__ A, const _Float16* __restrict__ Bt,
                       float* __restrict__ C, int M, int Kpad, int Npad, int tilesN) {
  int wid  = (blockIdx.x * blockDim.x + threadIdx.x) >> 5;
  int lane = threadIdx.x & 31;
  int tm = wid / tilesN, tn = wid - tm * tilesN;
  if (tm >= ((M + 15) >> 4)) return;
  int g = lane >> 4, ln = lane & 15;
  const _Float16* Arow = A + (long)(tm * 16 + ln) * Kpad;
  const _Float16* Bbase = Bt + (long)(tn * (16 * NSUB) + ln) * Kpad + 16 * g;
  v8f acc[NSUB] = {};
  for (int k0 = 0; k0 < Kpad; k0 += 32) {
    __builtin_prefetch(Arow + k0 + 64, 0, 3);
    HF a;
    const U128* pa = (const U128*)(Arow + k0 + 16 * g);  // A: K = e + 16g
    a.q[0] = pa[0];
    a.q[1] = pa[1];
#pragma unroll
    for (int j = 0; j < NSUB; ++j) {
      HF b;                                              // B: lane=N, K = e + 16g
      const U128* pb = (const U128*)(Bbase + (long)(j * 16) * Kpad + k0);
      b.q[0] = pb[0];
      b.q[1] = pb[1];
      acc[j] = __builtin_amdgcn_wmma_f32_16x16x32_f16(false, a.h, false, b.h,
                                                      (short)0, acc[j], false, false);
    }
  }
#pragma unroll
  for (int j = 0; j < NSUB; ++j) {
    long base = (long)(tm * 16 + g * 8) * Npad + tn * (16 * NSUB) + j * 16 + ln;
#pragma unroll
    for (int r = 0; r < 8; ++r) C[base + (long)r * Npad] = acc[j][r];
  }
}

// Implicit-im2col conv GEMM over [NF, Cin, 25, 5], SAME padding, KSxKS kernel.
// Rows = NF*125; C[row, Npad]. LOGPATH applies log(|x|+eps) BEFORE zero padding.
// Branch-free gather: unconditional load from clamped offset + mask-select.
template <int LOGPATH, int KS, int NSUB>
__global__ void k_conv_gemm(const float* __restrict__ X, const _Float16* __restrict__ Bt,
                            float* __restrict__ C, int NF, int Cin,
                            int K, int Kpad, int Npad, int tilesN) {
  int wid  = (blockIdx.x * blockDim.x + threadIdx.x) >> 5;
  int lane = threadIdx.x & 31;
  int tm = wid / tilesN, tn = wid - tm * tilesN;
  int M = NF * 125;
  if (tm * 16 >= M) return;
  int g = lane >> 4, ln = lane & 15;
  int row = tm * 16 + ln;
  int n = row / 125, hw = row - n * 125;
  int h = hw / 5, w = hw - h * 5;
  const float* Xn = X + (long)n * Cin * 125;
  const _Float16* Bbase = Bt + (long)(tn * (16 * NSUB) + ln) * Kpad + 16 * g;
  v8f acc[NSUB] = {};
  for (int k0 = 0; k0 < Kpad; k0 += 32) {
    HF a;
    v16h av;
#pragma unroll
    for (int e = 0; e < 16; ++e) {
      int kk = k0 + e + 16 * g;
      int off;
      bool valid;
      if (KS == 3) {
        int ci = kk / 9;
        int r9 = kk - ci * 9;
        int ih = h + r9 / 3 - 1;
        int iw = w + (r9 % 3) - 1;
        valid = (kk < K) & ((unsigned)ih < 25u) & ((unsigned)iw < 5u);
        off = ci * 125 + ih * 5 + iw;
      } else {
        valid = (kk < K);
        off = kk * 125 + hw;
      }
      off = valid ? off : 0;              // clamped, always-legal address
      float v = Xn[off];                  // unconditional load
      if (LOGPATH) v = __logf(fabsf(v) + NALU_EPS);
      v = valid ? v : 0.0f;               // mask-select instead of branch
      av[e] = (_Float16)v;
    }
    a.h = av;
#pragma unroll
    for (int j = 0; j < NSUB; ++j) {
      HF b;
      const U128* pb = (const U128*)(Bbase + (long)(j * 16) * Kpad + k0);
      b.q[0] = pb[0];
      b.q[1] = pb[1];
      acc[j] = __builtin_amdgcn_wmma_f32_16x16x32_f16(false, a.h, false, b.h,
                                                      (short)0, acc[j], false, false);
    }
  }
#pragma unroll
  for (int j = 0; j < NSUB; ++j) {
    long base = (long)(tm * 16 + g * 8) * Npad + tn * (16 * NSUB) + j * 16 + ln;
#pragma unroll
    for (int r = 0; r < 8; ++r) C[base + (long)r * Npad] = acc[j][r];
  }
}

// ---------------------------------------------------------------- NALU / BN / misc
__global__ void k_nalu_combine_conv(const float* __restrict__ a, const float* __restrict__ m,
                                    const float* __restrict__ gg, float* __restrict__ dst,
                                    int NF, int Cout, int Npad, int Ctot, int coff) {
  long idx = (long)blockIdx.x * blockDim.x + threadIdx.x;
  long total = (long)NF * 125 * Cout;
  if (idx >= total) return;
  int c = (int)(idx % Cout);
  long row = idx / Cout;
  int n = (int)(row / 125), hw = (int)(row - (long)n * 125);
  long s = row * Npad + c;
  float gv = sigm_(gg[s]);
  float out = gv * a[s] + (1.0f - gv) * __expf(m[s]);
  dst[((long)n * Ctot + coff + c) * 125 + hw] = out;
}

__global__ void k_copy_ch(const float* __restrict__ src, float* __restrict__ dst,
                          int NF, int Cs, int Ctot) {
  long idx = (long)blockIdx.x * blockDim.x + threadIdx.x;
  long total = (long)NF * Cs * 125;
  if (idx >= total) return;
  int hw = (int)(idx % 125);
  long t = idx / 125;
  int c = (int)(t % Cs);
  int n = (int)(t / Cs);
  dst[((long)n * Ctot + c) * 125 + hw] = src[idx];
}

__global__ void k_bn_stats(const float* __restrict__ X, float* __restrict__ stats, int NF, int C) {
  int ch = blockIdx.y;
  long total = (long)NF * 125;
  float s = 0.0f, s2 = 0.0f;
  for (long i = (long)blockIdx.x * blockDim.x + threadIdx.x; i < total;
       i += (long)gridDim.x * blockDim.x) {
    long n = i / 125;
    int hw = (int)(i - n * 125);
    float v = X[(n * C + ch) * 125 + hw];
    s += v; s2 += v * v;
  }
  __shared__ float r1[256];
  __shared__ float r2[256];
  int t = threadIdx.x;
  r1[t] = s; r2[t] = s2;
  __syncthreads();
  for (int off = 128; off > 0; off >>= 1) {
    if (t < off) { r1[t] += r1[t + off]; r2[t] += r2[t + off]; }
    __syncthreads();
  }
  if (t == 0) {
    atomicAdd(&stats[ch * 2 + 0], r1[0]);
    atomicAdd(&stats[ch * 2 + 1], r2[0]);
  }
}

__global__ void k_bn_apply(float* __restrict__ X, const float* __restrict__ stats,
                           const float* __restrict__ gamma, const float* __restrict__ beta,
                           int NF, int C) {
  long idx = (long)blockIdx.x * blockDim.x + threadIdx.x;
  long total = (long)NF * C * 125;
  if (idx >= total) return;
  int c = (int)((idx / 125) % C);
  float cnt = (float)NF * 125.0f;
  float mu = stats[c * 2] / cnt;
  float var = stats[c * 2 + 1] / cnt - mu * mu;
  X[idx] = (X[idx] - mu) * rsqrtf(var + BN_EPS) * gamma[c] + beta[c];
}

__global__ void k_build_comb(const float* __restrict__ timing, const float* __restrict__ xo,
                             float* __restrict__ cf32, _Float16* __restrict__ cf16) {
  int idx = blockIdx.x * blockDim.x + threadIdx.x;
  if (idx >= 8192 * 224) return;
  int d = idx % 224, tb = idx / 224;
  float v = 0.0f;
  if (d < 96)       v = timing[tb * 96 + d];
  else if (d < 221) v = xo[tb * 125 + (d - 96)];
  cf16[idx] = (_Float16)v;
  if (d < 221) cf32[tb * 221 + d] = v;
}

__global__ void k_gru_gate(const float* __restrict__ gi, const float* __restrict__ gh,
                           const float* __restrict__ bih, const float* __restrict__ bhh,
                           const float* __restrict__ inp, float* __restrict__ h,
                           _Float16* __restrict__ hf16, int residual) {
  int idx = blockIdx.x * blockDim.x + threadIdx.x;
  if (idx >= 1024 * 221) return;
  int b = idx / 221, d = idx - b * 221;
  long gb = (long)b * 672;
  float ir = gi[gb + d]       + bih[d];
  float iz = gi[gb + 221 + d] + bih[221 + d];
  float in_ = gi[gb + 442 + d] + bih[442 + d];
  float hr = gh[gb + d]       + bhh[d];
  float hz = gh[gb + 221 + d] + bhh[221 + d];
  float hn = gh[gb + 442 + d] + bhh[442 + d];
  float r = sigm_(ir + hr);
  float z = sigm_(iz + hz);
  float nn = tanhf(in_ + r * hn);
  float hnew = (1.0f - z) * nn + z * h[idx];
  if (residual) hnew += inp[idx];
  h[idx] = hnew;
  hf16[b * 224 + d] = (_Float16)hnew;
}

__global__ void k_nalu_dense(const float* __restrict__ a, const float* __restrict__ m,
                             const float* __restrict__ gg, float* __restrict__ out,
                             _Float16* __restrict__ of16, _Float16* __restrict__ olog16,
                             int Dout, int Npad, int Kp16) {
  int idx = blockIdx.x * blockDim.x + threadIdx.x;
  if (idx >= 1024 * Dout) return;
  int b = idx / Dout, d = idx - b * Dout;
  long s = (long)b * Npad + d;
  float gv = sigm_(gg[s]);
  float o = gv * a[s] + (1.0f - gv) * __expf(m[s]);
  out[idx] = o;
  if (of16)   of16[b * Kp16 + d] = (_Float16)o;
  if (olog16) olog16[b * Kp16 + d] = (_Float16)__logf(fabsf(o) + NALU_EPS);
}

// ---------------------------------------------------------------- host
static inline int cdiv(long a, long b) { return (int)((a + b - 1) / b); }

extern "C" void kernel_launch(void* const* d_in, const int* in_sizes, int n_in,
                              void* d_out, int out_size, void* d_ws, size_t ws_size,
                              hipStream_t stream) {
  (void)in_sizes; (void)n_in; (void)out_size; (void)ws_size;
  const float* x      = (const float*)d_in[0];
  const float* timing = (const float*)d_in[1];

  char* wp = (char*)d_ws;
  auto alloc = [&](size_t bytes) -> void* {
    void* p = (void*)wp;
    wp += (bytes + 255) & ~(size_t)255;
    return p;
  };

  _Float16* cW1 = (_Float16*)alloc(16 * 32 * 2);   _Float16* cG1 = (_Float16*)alloc(16 * 32 * 2);
  _Float16* cW2 = (_Float16*)alloc(32 * 160 * 2);  _Float16* cG2 = (_Float16*)alloc(32 * 160 * 2);
  _Float16* cW3 = (_Float16*)alloc(64 * 448 * 2);  _Float16* cG3 = (_Float16*)alloc(64 * 448 * 2);
  _Float16* cW4 = (_Float16*)alloc(16 * 128 * 2);  _Float16* cG4 = (_Float16*)alloc(16 * 128 * 2);
  _Float16* gWih = (_Float16*)alloc((size_t)5 * 672 * 224 * 2);
  _Float16* gWhh = (_Float16*)alloc((size_t)5 * 672 * 224 * 2);
  _Float16* n1W = (_Float16*)alloc(128 * 224 * 2); _Float16* n1G = (_Float16*)alloc(128 * 224 * 2);
  _Float16* n2W = (_Float16*)alloc(128 * 128 * 2); _Float16* n2G = (_Float16*)alloc(128 * 128 * 2);
  float* bufA = (float*)alloc((size_t)1024000 * 64 * 4);
  float* bufM = (float*)alloc((size_t)1024000 * 64 * 4);
  float* bufG = (float*)alloc((size_t)1024000 * 64 * 4);
  float* x1 = (float*)alloc((size_t)8192 * 16 * 125 * 4);
  float* x3 = (float*)alloc((size_t)8192 * 48 * 125 * 4);
  float* x4 = (float*)alloc((size_t)8192 * 112 * 125 * 4);
  float* xo = (float*)alloc((size_t)8192 * 125 * 4);
  float* stats = (float*)alloc(256 * 4);
  float* combf = (float*)alloc((size_t)8192 * 221 * 4);
  _Float16* combh = (_Float16*)alloc((size_t)8192 * 224 * 2);
  float* hM = (float*)alloc((size_t)6 * 1024 * 221 * 4);
  _Float16* hH = (_Float16*)alloc((size_t)6 * 1024 * 224 * 2);
  float* gi = (float*)alloc((size_t)1024 * 672 * 4);
  float* gh = (float*)alloc((size_t)1024 * 672 * 4);
  _Float16* loghH = (_Float16*)alloc((size_t)1024 * 224 * 2);
  float* da = (float*)alloc((size_t)1024 * 128 * 4);
  float* dm = (float*)alloc((size_t)1024 * 128 * 4);
  float* dg = (float*)alloc((size_t)1024 * 128 * 4);
  float* o1 = (float*)alloc((size_t)1024 * 125 * 4);
  _Float16* o1H  = (_Float16*)alloc((size_t)1024 * 128 * 2);
  _Float16* o1LH = (_Float16*)alloc((size_t)1024 * 128 * 2);

  // ---- weight packing
  auto packw = [&](int iw, int im, int ig, _Float16* Wd, _Float16* Gd, int O, int K, int Op, int Kp) {
    k_pack_nalu_w<<<cdiv((long)Op * Kp, 256), 256, 0, stream>>>(
        (const float*)d_in[iw], (const float*)d_in[im], (const float*)d_in[ig], Wd, Gd, O, K, Op, Kp);
  };
  packw(2, 3, 4,    cW1, cG1, 15, 9,   16, 32);
  packw(7, 8, 9,    cW2, cG2, 32, 144, 32, 160);
  packw(12, 13, 14, cW3, cG3, 64, 432, 64, 448);
  packw(17, 18, 19, cW4, cG4, 1, 112,  16, 128);
  packw(26, 27, 28, n1W, n1G, 125, 221, 128, 224);
  packw(29, 30, 31, n2W, n2G, 125, 125, 128, 128);
  for (int j = 0; j < 5; ++j) {
    k_pack_f16<<<cdiv(672L * 224, 256), 256, 0, stream>>>(
        (const float*)d_in[22] + (size_t)j * 663 * 221, gWih + (size_t)j * 672 * 224,
        663, 221, 672, 224, 0);
    k_pack_f16<<<cdiv(672L * 224, 256), 256, 0, stream>>>(
        (const float*)d_in[23] + (size_t)j * 663 * 221, gWhh + (size_t)j * 672 * 224,
        663, 221, 672, 224, 0);
  }

  // ---- conv stages (NSUB chosen so Npad == 16*NSUB*tilesN exactly)
  // conv1: 1 -> 15 (Npad 16, NSUB 1), concat -> 16, BN
  {
    int tn = 1, blocks = cdiv(64000L * tn, 4);
    k_conv_gemm<0, 3, 1><<<blocks, 128, 0, stream>>>(x, cW1, bufA, 8192, 1, 9, 32, 16, tn);
    k_conv_gemm<1, 3, 1><<<blocks, 128, 0, stream>>>(x, cW1, bufM, 8192, 1, 9, 32, 16, tn);
    k_conv_gemm<0, 3, 1><<<blocks, 128, 0, stream>>>(x, cG1, bufG, 8192, 1, 9, 32, 16, tn);
  }
  k_copy_ch<<<cdiv(8192L * 1 * 125, 256), 256, 0, stream>>>(x, x1, 8192, 1, 16);
  k_nalu_combine_conv<<<cdiv(1024000L * 15, 256), 256, 0, stream>>>(bufA, bufM, bufG, x1, 8192, 15, 16, 16, 1);
  {
    k_zero<<<1, 256, 0, stream>>>((unsigned int*)stats, 256);
    dim3 gs(64, 16);
    k_bn_stats<<<gs, 256, 0, stream>>>(x1, stats, 8192, 16);
    k_bn_apply<<<cdiv(8192L * 16 * 125, 256), 256, 0, stream>>>(
        x1, stats, (const float*)d_in[5], (const float*)d_in[6], 8192, 16);
  }

  // conv2: 16 -> 32 (Npad 32, NSUB 2), concat -> 48, BN
  {
    int tn = 1, blocks = cdiv(64000L * tn, 4);
    k_conv_gemm<0, 3, 2><<<blocks, 128, 0, stream>>>(x1, cW2, bufA, 8192, 16, 144, 160, 32, tn);
    k_conv_gemm<1, 3, 2><<<blocks, 128, 0, stream>>>(x1, cW2, bufM, 8192, 16, 144, 160, 32, tn);
    k_conv_gemm<0, 3, 2><<<blocks, 128, 0, stream>>>(x1, cG2, bufG, 8192, 16, 144, 160, 32, tn);
  }
  k_copy_ch<<<cdiv(8192L * 16 * 125, 256), 256, 0, stream>>>(x1, x3, 8192, 16, 48);
  k_nalu_combine_conv<<<cdiv(1024000L * 32, 256), 256, 0, stream>>>(bufA, bufM, bufG, x3, 8192, 32, 32, 48, 16);
  {
    k_zero<<<1, 256, 0, stream>>>((unsigned int*)stats, 256);
    dim3 gs(64, 48);
    k_bn_stats<<<gs, 256, 0, stream>>>(x3, stats, 8192, 48);
    k_bn_apply<<<cdiv(8192L * 48 * 125, 256), 256, 0, stream>>>(
        x3, stats, (const float*)d_in[10], (const float*)d_in[11], 8192, 48);
  }

  // conv3: 48 -> 64 (Npad 64, NSUB 4), concat -> 112, BN
  {
    int tn = 1, blocks = cdiv(64000L * tn, 4);
    k_conv_gemm<0, 3, 4><<<blocks, 128, 0, stream>>>(x3, cW3, bufA, 8192, 48, 432, 448, 64, tn);
    k_conv_gemm<1, 3, 4><<<blocks, 128, 0, stream>>>(x3, cW3, bufM, 8192, 48, 432, 448, 64, tn);
    k_conv_gemm<0, 3, 4><<<blocks, 128, 0, stream>>>(x3, cG3, bufG, 8192, 48, 432, 448, 64, tn);
  }
  k_copy_ch<<<cdiv(8192L * 48 * 125, 256), 256, 0, stream>>>(x3, x4, 8192, 48, 112);
  k_nalu_combine_conv<<<cdiv(1024000L * 64, 256), 256, 0, stream>>>(bufA, bufM, bufG, x4, 8192, 64, 64, 112, 48);
  {
    k_zero<<<1, 256, 0, stream>>>((unsigned int*)stats, 256);
    dim3 gs(64, 112);
    k_bn_stats<<<gs, 256, 0, stream>>>(x4, stats, 8192, 112);
    k_bn_apply<<<cdiv(8192L * 112 * 125, 256), 256, 0, stream>>>(
        x4, stats, (const float*)d_in[15], (const float*)d_in[16], 8192, 112);
  }

  // conv4: 112 -> 1 (1x1, Npad 16, NSUB 1), BN
  {
    int tn = 1, blocks = cdiv(64000L, 4);
    k_conv_gemm<0, 1, 1><<<blocks, 128, 0, stream>>>(x4, cW4, bufA, 8192, 112, 112, 128, 16, tn);
    k_conv_gemm<1, 1, 1><<<blocks, 128, 0, stream>>>(x4, cW4, bufM, 8192, 112, 112, 128, 16, tn);
    k_conv_gemm<0, 1, 1><<<blocks, 128, 0, stream>>>(x4, cG4, bufG, 8192, 112, 112, 128, 16, tn);
  }
  k_nalu_combine_conv<<<cdiv(1024000L, 256), 256, 0, stream>>>(bufA, bufM, bufG, xo, 8192, 1, 16, 1, 0);
  {
    k_zero<<<1, 256, 0, stream>>>((unsigned int*)stats, 256);
    dim3 gs(64, 1);
    k_bn_stats<<<gs, 256, 0, stream>>>(xo, stats, 8192, 1);
    k_bn_apply<<<cdiv(8192L * 125, 256), 256, 0, stream>>>(
        xo, stats, (const float*)d_in[20], (const float*)d_in[21], 8192, 1);
  }

  // ---- recurrent part
  k_build_comb<<<cdiv(8192L * 224, 256), 256, 0, stream>>>(timing, xo, combf, combh);
  k_zero<<<2048, 256, 0, stream>>>((unsigned int*)hM, 6L * 1024 * 221);
  k_zero<<<2048, 256, 0, stream>>>((unsigned int*)hH, 6L * 1024 * 224 / 2);
  k_zero<<<256, 256, 0, stream>>>((unsigned int*)o1H, 1024L * 128 / 2);
  k_zero<<<256, 256, 0, stream>>>((unsigned int*)o1LH, 1024L * 128 / 2);

  // GRU GEMMs: Npad 672 = 21 * 32 -> NSUB 2, tilesN 21
  auto gemm672 = [&](const _Float16* A, const _Float16* Bt, float* Cd) {
    int tn = 21;
    int blocks = cdiv(64L * tn, 4);
    k_gemm<2><<<blocks, 128, 0, stream>>>(A, Bt, Cd, 1024, 224, 672, tn);
  };
  // dense NALU GEMMs: Npad 128 -> NSUB 4, tilesN 2
  auto gemm128 = [&](const _Float16* A, const _Float16* Bt, float* Cd, int Kp) {
    int tn = 2;
    int blocks = cdiv(64L * tn, 4);
    k_gemm<4><<<blocks, 128, 0, stream>>>(A, Bt, Cd, 1024, Kp, 128, tn);
  };

  const float* bih_all = (const float*)d_in[24];
  const float* bhh_all = (const float*)d_in[25];
  for (int t = 0; t < 8; ++t) {
    for (int c = 0; c < 6; ++c) {
      int j = (c < 5) ? c : 4;
      const _Float16* inpH = (c == 0) ? combh + (size_t)t * 1024 * 224
                                      : hH + (size_t)(c - 1) * 1024 * 224;
      const float* inpF = (c == 0) ? combf + (size_t)t * 1024 * 221
                                   : hM + (size_t)(c - 1) * 1024 * 221;
      gemm672(inpH, gWih + (size_t)j * 672 * 224, gi);
      gemm672(hH + (size_t)c * 1024 * 224, gWhh + (size_t)j * 672 * 224, gh);
      k_gru_gate<<<cdiv(1024L * 221, 256), 256, 0, stream>>>(
          gi, gh, bih_all + (size_t)j * 663, bhh_all + (size_t)j * 663,
          inpF, hM + (size_t)c * 1024 * 221, hH + (size_t)c * 1024 * 224, (t < 7) ? 1 : 0);
    }
  }

  // ---- final NALU x2
  const _Float16* h5H = hH + (size_t)5 * 1024 * 224;
  k_pack_f16<<<cdiv(1024L * 224, 256), 256, 0, stream>>>(
      hM + (size_t)5 * 1024 * 221, loghH, 1024, 221, 1024, 224, 1);
  gemm128(h5H,   n1W, da, 224);
  gemm128(loghH, n1W, dm, 224);
  gemm128(h5H,   n1G, dg, 224);
  k_nalu_dense<<<cdiv(1024L * 125, 256), 256, 0, stream>>>(da, dm, dg, o1, o1H, o1LH, 125, 128, 128);
  gemm128(o1H,  n2W, da, 128);
  gemm128(o1LH, n2W, dm, 128);
  gemm128(o1H,  n2G, dg, 128);
  k_nalu_dense<<<cdiv(1024L * 125, 256), 256, 0, stream>>>(da, dm, dg, (float*)d_out,
                                                           nullptr, nullptr, 125, 128, 128);
}
